// SelfAttention_46316927320362
// MI455X (gfx1250) — matrix operand order repaired and checked by reading, not compiled, not run on hardware
//
#include <hip/hip_runtime.h>
#include <stdint.h>

#define EMB 1024
#define HEADS 16
#define HIDDEN 64
#define BATCH 2
#define SEQ 2048
#define ROWM (HEADS * HIDDEN)   // 1024, row stride of all (n, h*d) tensors

typedef __attribute__((ext_vector_type(16))) __bf16        v16bf;
typedef __attribute__((ext_vector_type(8)))  float         v8f;
typedef __attribute__((ext_vector_type(4)))  unsigned int  u32x4;
typedef __attribute__((ext_vector_type(4)))  int           v4i;

typedef v4i __attribute__((address_space(1)))* gv4i_p;  // global int4*
typedef v4i __attribute__((address_space(3)))* lv4i_p;  // LDS int4*

union FragU { unsigned int u[8]; v16bf v; };
union U4S8 { u32x4 q; unsigned short s[8]; };

#if defined(__has_builtin)
#if __has_builtin(__builtin_amdgcn_global_load_async_to_lds_b128)
#define HAVE_ASYNC_LDS 1
#else
#define HAVE_ASYNC_LDS 0
#endif
#else
#define HAVE_ASYNC_LDS 0
#endif

// 16-byte global -> LDS copy; async (ASYNCcnt-tracked, no VGPR round trip)
// when the gfx1250 builtin is available, else through VGPRs.
static __device__ __forceinline__ void copy16B_to_lds(const unsigned short* gsrc,
                                                      unsigned short* ldst) {
#if HAVE_ASYNC_LDS
  __builtin_amdgcn_global_load_async_to_lds_b128(
      (gv4i_p)(void*)gsrc, (lv4i_p)(void*)ldst, 0, 0);
#else
  *(u32x4*)ldst = *(const u32x4*)gsrc;
#endif
}

static __device__ __forceinline__ void async_wait0() {
#if HAVE_ASYNC_LDS
#if __has_builtin(__builtin_amdgcn_s_wait_asynccnt)
  __builtin_amdgcn_s_wait_asynccnt(0);
#else
  asm volatile("s_wait_asynccnt 0" ::: "memory");
#endif
#endif
}

static __device__ __forceinline__ unsigned short f32_to_bf16(float f) {
  unsigned int u = __float_as_uint(f);
  u += 0x7FFFu + ((u >> 16) & 1u);   // round-to-nearest-even
  return (unsigned short)(u >> 16);
}

static __device__ __forceinline__ float redmax16(float v) {
  v = fmaxf(v, __shfl_xor(v, 1, 32));
  v = fmaxf(v, __shfl_xor(v, 2, 32));
  v = fmaxf(v, __shfl_xor(v, 4, 32));
  v = fmaxf(v, __shfl_xor(v, 8, 32));
  return v;
}
static __device__ __forceinline__ float redsum16(float v) {
  v += __shfl_xor(v, 1, 32);
  v += __shfl_xor(v, 2, 32);
  v += __shfl_xor(v, 4, 32);
  v += __shfl_xor(v, 8, 32);
  return v;
}

// K-offset of A-fragment register i for the 16x32 bf16 A layout
// (lanes 0-15: K {0..7,16..23}; lanes 16-31: K {8..15,24..31})
static __device__ __forceinline__ int a_koff(int i, int hi) {
  return (i < 4) ? (hi * 8 + 2 * i) : (16 + hi * 8 + 2 * (i - 4));
}

// ---------------------------------------------------------------- fp32->bf16
__global__ void cvt_f32_bf16_kernel(const float* __restrict__ in,
                                    unsigned short* __restrict__ out, int n) {
  int i = blockIdx.x * blockDim.x + threadIdx.x;
  if (i < n) out[i] = f32_to_bf16(in[i]);
}

// ------------------------------------------------------- generic bf16 GEMM
// C[m,n] = scale * sum_k A[m,k]*B[k,n] (+ bias[n]); A,B bf16 row-major.
// Block: 128 threads = 4 waves; block computes 64x64 of C.  Each wave owns
// 16 M-rows and sweeps all four 16-wide N slices: 4 WMMAs per wave per
// K-step (A-frag reused across the four B-frags).  grid = (N/64, M/64).
__global__ __launch_bounds__(128) void gemm_bf16_wmma_kernel(
    const unsigned short* __restrict__ A, const unsigned short* __restrict__ B,
    unsigned short* __restrict__ Cb, float* __restrict__ Cf,
    const float* __restrict__ bias,
    int Kk, int lda, int ldb, int ldc, float scale) {
  const int tid  = threadIdx.x;
  const int wave = tid >> 5;
  const int lane = tid & 31;
  const int ln   = lane & 15;
  const int hi   = lane >> 4;
  const int m0   = blockIdx.y * 64;
  const int n0   = blockIdx.x * 64;

  __shared__ unsigned short As[64][40];    // 64 rows x 32 k, padded
  __shared__ unsigned short Bts[64][40];   // transposed: [n][k], padded

  v8f acc[4];
#pragma unroll
  for (int t = 0; t < 4; ++t)
#pragma unroll
    for (int r = 0; r < 8; ++r) acc[t][r] = 0.0f;

  const int arow = tid >> 1;          // A stage: 64 rows x (2 x 16 elems)
  const int aseg = (tid & 1) * 16;
  const int bk   = tid >> 2;          // B stage: 32 k-rows x (4 x 16 n)
  const int bnp  = (tid & 3) * 16;

  for (int k0 = 0; k0 < Kk; k0 += 32) {
    __syncthreads();
    {  // stage A 64x32 (async 16B global->LDS copies)
      const unsigned short* src = A + (size_t)(m0 + arow) * lda + (k0 + aseg);
      copy16B_to_lds(src,     &As[arow][aseg]);
      copy16B_to_lds(src + 8, &As[arow][aseg + 8]);
    }
    {  // stage B 32x64 transposed into [n][k]
      const unsigned short* src = B + (size_t)(k0 + bk) * ldb + (n0 + bnp);
      U4S8 d0, d1;
      d0.q = *(const u32x4*)src;
      d1.q = *(const u32x4*)(src + 8);
#pragma unroll
      for (int j = 0; j < 8; ++j) Bts[bnp + j][bk]     = d0.s[j];
#pragma unroll
      for (int j = 0; j < 8; ++j) Bts[bnp + 8 + j][bk] = d1.s[j];
    }
    async_wait0();
    __syncthreads();

    FragU fa;
    const unsigned short* arow_p = &As[wave * 16 + ln][0];
#pragma unroll
    for (int i = 0; i < 8; ++i)
      fa.u[i] = *(const unsigned int*)(arow_p + a_koff(i, hi));
#pragma unroll
    for (int t = 0; t < 4; ++t) {
      FragU fb;
      const unsigned short* brow_p = &Bts[16 * t + ln][0];
#pragma unroll
      for (int i = 0; i < 8; ++i)
        fb.u[i] = *(const unsigned int*)(brow_p + a_koff(i, hi));
      acc[t] = __builtin_amdgcn_wmma_f32_16x16x32_bf16(false, fa.v, false, fb.v,
                                                       (short)0, acc[t],
                                                       false, false);
    }
  }

#pragma unroll
  for (int t = 0; t < 4; ++t) {
    const int n = n0 + 16 * t + ln;
    const float bv = bias ? bias[n] : 0.0f;
#pragma unroll
    for (int r = 0; r < 8; ++r) {
      const int m = m0 + wave * 16 + r + 8 * hi;  // C layout: reg r -> row r+8*hi
      float v = acc[t][r] * scale + bv;
      if (Cb) Cb[(size_t)m * ldc + n] = f32_to_bf16(v);
      else    Cf[(size_t)m * ldc + n] = v;
    }
  }
}

// ------------------------------------------------- flash attention (causal)
// Block = (batch b, head h, 64 query rows); 4 waves, each owns 16 q-rows.
// Q/K already carry the 1/sqrt(hd) scale from the projection.
__global__ __launch_bounds__(128) void attn_fwd_kernel(
    const unsigned short* __restrict__ Q, const unsigned short* __restrict__ K,
    const unsigned short* __restrict__ V, unsigned short* __restrict__ O) {
  const int tid  = threadIdx.x;
  const int wave = tid >> 5;
  const int lane = tid & 31;
  const int ln   = lane & 15;
  const int hi   = lane >> 4;
  const int qblk = blockIdx.x;
  const int h    = blockIdx.y;
  const int b    = blockIdx.z;

  __shared__ unsigned short Ks[32][72];      // key rows, natural [krow][d]
  __shared__ unsigned short Vt[64][40];      // values transposed [d][krow]
  __shared__ unsigned short Ps[4][16][40];   // per-wave P tile C->A relayout

  const size_t headBase = (size_t)b * SEQ * ROWM + (size_t)h * HIDDEN;
  const int q0 = qblk * 64;
  const int qb = q0 + wave * 16;

  // Q fragments: 16 rows x 64 d as two 16x32 A-frags (kept in VGPRs)
  FragU qf[2];
  {
    const unsigned short* qrow = Q + headBase + (size_t)(qb + ln) * ROWM;
#pragma unroll
    for (int s = 0; s < 2; ++s)
#pragma unroll
      for (int i = 0; i < 8; ++i)
        qf[s].u[i] = *(const unsigned int*)(qrow + 32 * s + a_koff(i, hi));
  }

  float mrow[8], lrow[8];
  v8f oacc[4];
#pragma unroll
  for (int r = 0; r < 8; ++r) { mrow[r] = -1e30f; lrow[r] = 0.0f; }
#pragma unroll
  for (int t = 0; t < 4; ++t)
#pragma unroll
    for (int r = 0; r < 8; ++r) oacc[t][r] = 0.0f;

  const unsigned short* Kh = K + headBase;
  const unsigned short* Vh = V + headBase;
  const int ldrow = tid >> 2;
  const int ldseg = (tid & 3) * 16;

  const int nkb = q0 / 32 + 2;   // uniform across the 4 waves (mask handles rest)
  for (int jb = 0; jb < nkb; ++jb) {
    const int kj = jb * 32;
    __syncthreads();
    {  // stage K block 32x64 (async 16B global->LDS copies)
      const unsigned short* src = Kh + (size_t)(kj + ldrow) * ROWM + ldseg;
      copy16B_to_lds(src,     &Ks[ldrow][ldseg]);
      copy16B_to_lds(src + 8, &Ks[ldrow][ldseg + 8]);
    }
    {  // stage V block transposed -> Vt[d][k]
      const unsigned short* src = Vh + (size_t)(kj + ldrow) * ROWM + ldseg;
      U4S8 d0, d1;
      d0.q = *(const u32x4*)src;
      d1.q = *(const u32x4*)(src + 8);
#pragma unroll
      for (int j = 0; j < 8; ++j) Vt[ldseg + j][ldrow]     = d0.s[j];
#pragma unroll
      for (int j = 0; j < 8; ++j) Vt[ldseg + 8 + j][ldrow] = d1.s[j];
    }
    async_wait0();
    __syncthreads();

    // S = Q K^T : two 16x16 C tiles across 32 key rows
    v8f sc[2];
#pragma unroll
    for (int t = 0; t < 2; ++t) {
#pragma unroll
      for (int r = 0; r < 8; ++r) sc[t][r] = 0.0f;
#pragma unroll
      for (int s = 0; s < 2; ++s) {
        FragU fb;
        const unsigned short* krow = &Ks[16 * t + ln][0];   // B column = K row
#pragma unroll
        for (int i = 0; i < 8; ++i)
          fb.u[i] = *(const unsigned int*)(krow + 32 * s + a_koff(i, hi));
        sc[t] = __builtin_amdgcn_wmma_f32_16x16x32_bf16(false, qf[s].v, false,
                                                        fb.v, (short)0, sc[t],
                                                        false, false);
      }
    }

    // causal mask + row max
    float rmax[8];
#pragma unroll
    for (int r = 0; r < 8; ++r) {
      const int qg = qb + r + 8 * hi;
#pragma unroll
      for (int t = 0; t < 2; ++t) {
        const int kg = kj + 16 * t + ln;
        if (kg > qg) sc[t][r] = -1e30f;
      }
      rmax[r] = redmax16(fmaxf(sc[0][r], sc[1][r]));
    }

    // online softmax rescale
    float corr[8];
#pragma unroll
    for (int r = 0; r < 8; ++r) {
      const float mnew = fmaxf(mrow[r], rmax[r]);
      corr[r] = __expf(mrow[r] - mnew);
      mrow[r] = mnew;
    }
#pragma unroll
    for (int t = 0; t < 4; ++t)
#pragma unroll
      for (int r = 0; r < 8; ++r) oacc[t][r] *= corr[r];

    // P = exp(S - m), write to per-wave LDS (C-layout -> A-layout relayout)
#pragma unroll
    for (int r = 0; r < 8; ++r) {
      const float p0 = __expf(sc[0][r] - mrow[r]);
      const float p1 = __expf(sc[1][r] - mrow[r]);
      Ps[wave][r + 8 * hi][ln]      = f32_to_bf16(p0);
      Ps[wave][r + 8 * hi][ln + 16] = f32_to_bf16(p1);
      lrow[r] = lrow[r] * corr[r] + redsum16(p0 + p1);
    }
    __syncthreads();

    // O += P V : one A-frag (16x32), four d-slices of 16
    FragU fp;
    const unsigned short* prow = &Ps[wave][ln][0];
#pragma unroll
    for (int i = 0; i < 8; ++i)
      fp.u[i] = *(const unsigned int*)(prow + a_koff(i, hi));
#pragma unroll
    for (int t = 0; t < 4; ++t) {
      FragU fv;
      const unsigned short* vrow = &Vt[16 * t + ln][0];   // B column = Vt row
#pragma unroll
      for (int i = 0; i < 8; ++i)
        fv.u[i] = *(const unsigned int*)(vrow + a_koff(i, hi));
      oacc[t] = __builtin_amdgcn_wmma_f32_16x16x32_bf16(false, fp.v, false,
                                                        fv.v, (short)0, oacc[t],
                                                        false, false);
    }
  }

  // normalize and store bf16
#pragma unroll
  for (int r = 0; r < 8; ++r) {
    const float inv = 1.0f / lrow[r];
    const int qg = qb + r + 8 * hi;
    unsigned short* orow = O + headBase + (size_t)qg * ROWM;
#pragma unroll
    for (int t = 0; t < 4; ++t)
      orow[16 * t + ln] = f32_to_bf16(oacc[t][r] * inv);
  }
}

// --------------------------------------------------------------------------
extern "C" void kernel_launch(void* const* d_in, const int* in_sizes, int n_in,
                              void* d_out, int out_size, void* d_ws, size_t ws_size,
                              hipStream_t stream) {
  (void)in_sizes; (void)n_in; (void)out_size; (void)ws_size;

  const float* x   = (const float*)d_in[0];
  const float* w_q = (const float*)d_in[1];
  const float* w_k = (const float*)d_in[2];
  const float* w_v = (const float*)d_in[3];
  const float* w_u = (const float*)d_in[4];
  const float* b_u = (const float*)d_in[5];
  float* out = (float*)d_out;

  const int rows = BATCH * SEQ;        // 4096
  const int xN   = rows * EMB;         // 4,194,304
  const int wN   = EMB * ROWM;         // 1,048,576

  // workspace layout (bf16 = ushort), ~48 MB total
  unsigned short* ws  = (unsigned short*)d_ws;
  unsigned short* xb  = ws;
  unsigned short* wqb = xb  + xN;
  unsigned short* wkb = wqb + wN;
  unsigned short* wvb = wkb + wN;
  unsigned short* wub = wvb + wN;
  unsigned short* Qb  = wub + wN;
  unsigned short* Kb  = Qb  + xN;
  unsigned short* Vb  = Kb  + xN;
  unsigned short* Ab  = Vb  + xN;

  // fp32 -> bf16
  cvt_f32_bf16_kernel<<<dim3((xN + 255) / 256), dim3(256), 0, stream>>>(x, xb, xN);
  cvt_f32_bf16_kernel<<<dim3((wN + 255) / 256), dim3(256), 0, stream>>>(w_q, wqb, wN);
  cvt_f32_bf16_kernel<<<dim3((wN + 255) / 256), dim3(256), 0, stream>>>(w_k, wkb, wN);
  cvt_f32_bf16_kernel<<<dim3((wN + 255) / 256), dim3(256), 0, stream>>>(w_v, wvb, wN);
  cvt_f32_bf16_kernel<<<dim3((wN + 255) / 256), dim3(256), 0, stream>>>(w_u, wub, wN);

  // QKV projections (scale 1/sqrt(64) folded into Q and K, as in reference)
  const dim3 ggrid(ROWM / 64, rows / 64);   // (16, 64)
  const dim3 gblock(128);
  gemm_bf16_wmma_kernel<<<ggrid, gblock, 0, stream>>>(
      xb, wqb, Qb, nullptr, nullptr, EMB, EMB, ROWM, ROWM, 0.125f);
  gemm_bf16_wmma_kernel<<<ggrid, gblock, 0, stream>>>(
      xb, wkb, Kb, nullptr, nullptr, EMB, EMB, ROWM, ROWM, 0.125f);
  gemm_bf16_wmma_kernel<<<ggrid, gblock, 0, stream>>>(
      xb, wvb, Vb, nullptr, nullptr, EMB, EMB, ROWM, ROWM, 1.0f);

  // causal flash attention
  attn_fwd_kernel<<<dim3(SEQ / 64, HEADS, BATCH), dim3(128), 0, stream>>>(
      Qb, Kb, Vb, Ab);

  // output projection + bias -> fp32
  gemm_bf16_wmma_kernel<<<ggrid, gblock, 0, stream>>>(
      Ab, wub, nullptr, out, b_u, ROWM, ROWM, EMB, EMB, 1.0f);
}
